// CCAttention_vCR_31061203485396
// MI455X (gfx1250) — matrix operand order repaired and checked by reading, not compiled, not run on hardware
//
#include <hip/hip_runtime.h>
#include <hip/hip_bf16.h>

typedef __attribute__((ext_vector_type(16))) __bf16 v16bf;
typedef __attribute__((ext_vector_type(8)))  float  v8f;

#define HW   16384    // 128*128
#define WID  128
#define NOUT 33554432 // 4*512*16384
#define NP   8388608  // 4*128*16384

#define QT_S 136      // padded LDS strides (bf16 elems): 272B rows, 16B-multiple
#define A_S  136
#define K_S  136
#define V_S  136

// ---------------------------------------------------------------------------
// Kernel 1: grouped 1x1 conv as GEMM  Y[b,o,:] = W[o,:] @ X[b, g*256:(g+1)*256, :]
// Tile: 64(O) x 64(N) per WG, K=256 in steps of 32, bf16 WMMA, f32 accum.
// B-operand tile stored K-innermost (Xt[n][k]) so fragments are ds_load_b128.
// ---------------------------------------------------------------------------
__global__ __launch_bounds__(256)
void qkv_gemm(const float* __restrict__ x, const float* __restrict__ wmat,
              const float* __restrict__ bias, __bf16* __restrict__ out, int O) {
  const int bN = blockIdx.x * 64;       // HW tile base
  const int oT = blockIdx.y * 64;       // output-channel tile base
  const int bb = blockIdx.z;            // batch
  const int g  = (oT * 2) / O;          // conv group (constant per tile)
  const float* xg = x + ((size_t)bb * 512 + g * 256) * HW;

  __shared__ __bf16 Wl[64][40];         // A tile  [o][k]
  __shared__ __bf16 Xt[64][40];         // B tile  [n][k]  (k contiguous!)

  const int tid  = threadIdx.x;
  const int lane = tid & 31;
  const int wid  = tid >> 5;
  const int m    = lane & 15;
  const int half = lane >> 4;
  const int mt   = wid & 3;             // 16-row subtile
  const int nt0  = (wid >> 2) * 2;      // first of two 16-col subtiles

  const int rw = tid >> 2, cw = (tid & 3) * 8;   // W staging coords
  const int rx = tid >> 3, cx = (tid & 7) * 8;   // X staging coords

  v8f acc0 = {}; v8f acc1 = {};

  for (int k0 = 0; k0 < 256; k0 += 32) {
    __syncthreads();
    {
      const float4* wp = (const float4*)&wmat[(oT + rw) * 256 + k0 + cw];
      float4 w0 = wp[0], w1 = wp[1];
      __bf16* d = &Wl[rw][cw];
      d[0] = (__bf16)w0.x; d[1] = (__bf16)w0.y; d[2] = (__bf16)w0.z; d[3] = (__bf16)w0.w;
      d[4] = (__bf16)w1.x; d[5] = (__bf16)w1.y; d[6] = (__bf16)w1.z; d[7] = (__bf16)w1.w;

      const float4* xp = (const float4*)&xg[(size_t)(k0 + rx) * HW + bN + cx];
      float4 x0 = xp[0], x1 = xp[1];
      Xt[cx + 0][rx] = (__bf16)x0.x; Xt[cx + 1][rx] = (__bf16)x0.y;
      Xt[cx + 2][rx] = (__bf16)x0.z; Xt[cx + 3][rx] = (__bf16)x0.w;
      Xt[cx + 4][rx] = (__bf16)x1.x; Xt[cx + 5][rx] = (__bf16)x1.y;
      Xt[cx + 6][rx] = (__bf16)x1.z; Xt[cx + 7][rx] = (__bf16)x1.w;
      if (k0 < 224)                                  // prefetch next K slab
        __builtin_prefetch(&xg[(size_t)(k0 + 32 + rx) * HW + bN + cx], 0, 0);
    }
    __syncthreads();

    v16bf a, b0, b1;
#pragma unroll
    for (int e = 0; e < 16; ++e) {      // A frag: one contiguous 16-elem run
      int kk = e + (half << 4);
      a[e] = Wl[mt * 16 + m][kk];
    }
#pragma unroll
    for (int e = 0; e < 16; ++e) {      // B frag: one contiguous 16-elem run
      int kr = e + (half << 4);
      b0[e] = Xt[nt0 * 16 + m][kr];
      b1[e] = Xt[(nt0 + 1) * 16 + m][kr];
    }
    acc0 = __builtin_amdgcn_wmma_f32_16x16x32_bf16(false, a, false, b0, (short)0, acc0, false, false);
    acc1 = __builtin_amdgcn_wmma_f32_16x16x32_bf16(false, a, false, b1, (short)0, acc1, false, false);
  }

#pragma unroll
  for (int r = 0; r < 8; ++r) {
    int oo = oT + mt * 16 + r + 8 * half;
    int n0 = bN + nt0 * 16 + m;
    float bi = bias[oo];
    size_t base = ((size_t)bb * O + oo) * HW;
    out[base + n0]      = (__bf16)(acc0[r] + bi);
    out[base + n0 + 16] = (__bf16)(acc1[r] + bi);
  }
}

// ---------------------------------------------------------------------------
// Kernel 2: criss-cross attention, one WG per (dir, b*head, line).
// e = Q^T K (regs) -> softmax (half-wave shuffles) -> A (bf16, LDS) -> A V^T.
// All operand tiles stored so fragment gathers are contiguous b128 LDS loads.
// Dynamic LDS: Qt 34K + Kt 34K + V 17K + A 34K = 119 KB (2 WGs / 320KB WGP).
// ---------------------------------------------------------------------------
__global__ __launch_bounds__(256)
void cc_attention(const __bf16* __restrict__ q, const __bf16* __restrict__ k,
                  const __bf16* __restrict__ v, float* __restrict__ outH,
                  float* __restrict__ outW) {
  const int fixedc = blockIdx.x;        // fixed w (dir=0) or fixed h (dir=1)
  const int bh     = blockIdx.y;        // b*4 + head
  const int dir    = blockIdx.z;
  const int s    = (dir == 0) ? WID : 1;
  const int base = (dir == 0) ? fixedc : fixedc * WID;

  const __bf16* qg = q + (size_t)bh * 128 * HW;
  const __bf16* kg = k + (size_t)bh * 128 * HW;
  const __bf16* vg = v + (size_t)bh * 64 * HW;
  float* dst = ((dir == 0) ? outH : outW) + (size_t)bh * 64 * HW;

  extern __shared__ char smem[];
  __bf16* Qt = (__bf16*)smem;           // [i][c]   A of pass 1 (c contiguous)
  __bf16* Kt = Qt + 128 * QT_S;         // [j][c]   B of pass 1 (c contiguous)
  __bf16* Vl = Kt + 128 * K_S;          // [cv][j]  B of pass 2 (j contiguous)
  __bf16* Al = Vl + 64 * V_S;           // [i][j]   A of pass 2 (j contiguous)

  const int tid  = threadIdx.x;
  const int lane = tid & 31;
  const int wid  = tid >> 5;
  const int m    = lane & 15;
  const int half = lane >> 4;
  const int i0   = wid * 16;            // this wave's 16-row strip

  for (int l = tid; l < 16384; l += 256) {
    int c = l >> 7, p = l & 127;
    __bf16 qv = qg[(size_t)c * HW + p * s + base];
    __bf16 kv = kg[(size_t)c * HW + p * s + base];
    Qt[p * QT_S + c] = qv;
    Kt[p * K_S + c]  = kv;
  }
  for (int l = tid; l < 8192; l += 256) {
    int c = l >> 7, p = l & 127;
    Vl[c * V_S + p] = vg[(size_t)c * HW + p * s + base];
  }
  __syncthreads();

  // ---- pass 1: e = Q^T K, strip i0..i0+15 x all 128 j, K=128 over channels
  v8f acc[8];
#pragma unroll
  for (int jt = 0; jt < 8; ++jt) acc[jt] = (v8f){};

#pragma unroll
  for (int kc = 0; kc < 4; ++kc) {
    int k0 = kc * 32;
    v16bf a;
#pragma unroll
    for (int e = 0; e < 16; ++e) {
      int kk = e + (half << 4);
      a[e] = Qt[(i0 + m) * QT_S + k0 + kk];
    }
#pragma unroll
    for (int jt = 0; jt < 8; ++jt) {
      v16bf b;
#pragma unroll
      for (int e = 0; e < 16; ++e)
        b[e] = Kt[(jt * 16 + m) * K_S + k0 + e + (half << 4)];
      acc[jt] = __builtin_amdgcn_wmma_f32_16x16x32_bf16(false, a, false, b, (short)0, acc[jt], false, false);
    }
  }

  // ---- softmax over j (row = 16 lanes of one half-wave x 8 jt regs)
  const float scale = 0.08838834764831845f;  // 128^-0.5
#pragma unroll
  for (int r = 0; r < 8; ++r) {
    float mx = -3.4e38f;
#pragma unroll
    for (int jt = 0; jt < 8; ++jt) mx = fmaxf(mx, acc[jt][r]);
    mx = fmaxf(mx, __shfl_xor(mx, 8, 32));
    mx = fmaxf(mx, __shfl_xor(mx, 4, 32));
    mx = fmaxf(mx, __shfl_xor(mx, 2, 32));
    mx = fmaxf(mx, __shfl_xor(mx, 1, 32));
    float ex[8]; float sum = 0.f;
#pragma unroll
    for (int jt = 0; jt < 8; ++jt) { ex[jt] = __expf((acc[jt][r] - mx) * scale); sum += ex[jt]; }
    sum += __shfl_xor(sum, 8, 32);
    sum += __shfl_xor(sum, 4, 32);
    sum += __shfl_xor(sum, 2, 32);
    sum += __shfl_xor(sum, 1, 32);
    float inv = 1.f / sum;
    int i = i0 + r + 8 * half;          // C-matrix row for (r, half)
#pragma unroll
    for (int jt = 0; jt < 8; ++jt) Al[i * A_S + jt * 16 + m] = (__bf16)(ex[jt] * inv);
  }
  __syncthreads();

  // ---- pass 2: out[i, cv] = A @ V^T  (M=16 strip, N=64, K=128 over j)
  v8f acc2[4];
#pragma unroll
  for (int ct = 0; ct < 4; ++ct) acc2[ct] = (v8f){};
#pragma unroll
  for (int kc = 0; kc < 4; ++kc) {
    int k0 = kc * 32;
    v16bf a;
#pragma unroll
    for (int e = 0; e < 16; ++e) {
      int kk = e + (half << 4);
      a[e] = Al[(i0 + m) * A_S + k0 + kk];
    }
#pragma unroll
    for (int ct = 0; ct < 4; ++ct) {
      v16bf b;
#pragma unroll
      for (int e = 0; e < 16; ++e)
        b[e] = Vl[(ct * 16 + m) * V_S + k0 + e + (half << 4)];  // B[k=j][n=cv]
      acc2[ct] = __builtin_amdgcn_wmma_f32_16x16x32_bf16(false, a, false, b, (short)0, acc2[ct], false, false);
    }
  }

#pragma unroll
  for (int ct = 0; ct < 4; ++ct) {
#pragma unroll
    for (int r = 0; r < 8; ++r) {
      int p  = i0 + r + 8 * half;
      int cv = ct * 16 + m;
      dst[(size_t)cv * HW + p * s + base] = acc2[ct][r];
    }
  }
}

// ---------------------------------------------------------------------------
// Kernel 3: per-channel BN statistics over (B,H,W) for the 128 kept channels
// of each branch. stats[0..255] = mean (H then W), stats[256..511] = var.
// ---------------------------------------------------------------------------
__global__ __launch_bounds__(256)
void bn_stats(const float* __restrict__ outH, const float* __restrict__ outW,
              float* __restrict__ stats) {
  const int ch = blockIdx.x;            // 0..255
  const float* src = (ch < 128) ? outH : outW;
  const int cm = ch & 127;
  const int tid = threadIdx.x;
  float s = 0.f, ss = 0.f;
  for (int b = 0; b < 4; ++b)
    for (int pix = tid; pix < HW; pix += 256) {
      float v = src[((size_t)b * 256 + cm) * HW + pix];
      s += v; ss += v * v;
    }
  __shared__ float r1[256], r2[256];
  r1[tid] = s; r2[tid] = ss;
  __syncthreads();
  for (int o = 128; o; o >>= 1) {
    if (tid < o) { r1[tid] += r1[tid + o]; r2[tid] += r2[tid + o]; }
    __syncthreads();
  }
  if (tid == 0) {
    const float N = 65536.f;
    float mean = r1[0] / N;
    stats[ch]       = mean;
    stats[256 + ch] = r2[0] / N - mean * mean;
  }
}

// ---------------------------------------------------------------------------
// Kernel 4: out = gamma*relu(concat(out_H,out_W)) + x ; p0/p1 = BN slices.
// ---------------------------------------------------------------------------
__global__ __launch_bounds__(256)
void finalize(const float* __restrict__ x, const float* __restrict__ outH,
              const float* __restrict__ outW, const float* __restrict__ stats,
              const float* __restrict__ bnh_w, const float* __restrict__ bnh_b,
              const float* __restrict__ bnw_w, const float* __restrict__ bnw_b,
              const float* __restrict__ gamma, float* __restrict__ out) {
  const int idx = blockIdx.x * 256 + threadIdx.x;       // exact grid
  const int b   = idx >> 23;
  const int rem = idx & ((1 << 23) - 1);
  const int o   = rem >> 14;
  const int pix = rem & 16383;

  float val = (o < 256) ? outH[((size_t)b * 256 + o) * HW + pix]
                        : outW[((size_t)b * 256 + (o - 256)) * HW + pix];
  out[idx] = gamma[0] * fmaxf(val, 0.f) + x[idx];

  if (o < 128) {
    float mean = stats[o], var = stats[256 + o];
    out[NOUT + ((size_t)b * 128 + o) * HW + pix] =
        (val - mean) * rsqrtf(var + 1e-5f) * bnh_w[o] + bnh_b[o];
  } else if (o >= 256 && o < 384) {
    int ow = o - 256;
    float mean = stats[128 + ow], var = stats[384 + ow];
    out[NOUT + NP + ((size_t)b * 128 + ow) * HW + pix] =
        (val - mean) * rsqrtf(var + 1e-5f) * bnw_w[ow] + bnw_b[ow];
  }
}

// ---------------------------------------------------------------------------
extern "C" void kernel_launch(void* const* d_in, const int* in_sizes, int n_in,
                              void* d_out, int out_size, void* d_ws, size_t ws_size,
                              hipStream_t stream) {
  (void)in_sizes; (void)n_in; (void)out_size; (void)ws_size;
  const float* x     = (const float*)d_in[0];
  const float* wq    = (const float*)d_in[1];
  const float* bq    = (const float*)d_in[2];
  const float* wk    = (const float*)d_in[3];
  const float* bk    = (const float*)d_in[4];
  const float* wv    = (const float*)d_in[5];
  const float* bv    = (const float*)d_in[6];
  const float* bnh_w = (const float*)d_in[7];
  const float* bnh_b = (const float*)d_in[8];
  const float* bnw_w = (const float*)d_in[9];
  const float* bnw_b = (const float*)d_in[10];
  const float* gamma = (const float*)d_in[11];
  float* out = (float*)d_out;

  char* ws = (char*)d_ws;
  __bf16* qws = (__bf16*)(ws);                      //  64 MiB bf16 q
  __bf16* kws = (__bf16*)(ws + 67108864ull);        //  64 MiB bf16 k
  __bf16* vws = (__bf16*)(ws + 134217728ull);       //  32 MiB bf16 v
  float*  oH  = (float*)(ws + 167772160ull);        //  64 MiB f32 out_H
  float*  oW  = (float*)(ws + 234881024ull);        //  64 MiB f32 out_W
  float*  st  = (float*)(ws + 301989888ull);        //  2 KiB stats

  // 1) Q/K/V projections (bf16 WMMA GEMMs)
  qkv_gemm<<<dim3(256, 8, 4), 256, 0, stream>>>(x, wq, bq, qws, 512);
  qkv_gemm<<<dim3(256, 8, 4), 256, 0, stream>>>(x, wk, bk, kws, 512);
  qkv_gemm<<<dim3(256, 4, 4), 256, 0, stream>>>(x, wv, bv, vws, 256);

  // 2) Criss-cross attention (H and W directions), scores stay on-chip
  const int smem = (128 * QT_S + 128 * K_S + 64 * V_S + 128 * A_S) * 2;
  cc_attention<<<dim3(128, 16, 2), 256, smem, stream>>>(qws, kws, vws, oH, oW);

  // 3) BN statistics for kept channels
  bn_stats<<<dim3(256), 256, 0, stream>>>(oH, oW, st);

  // 4) Fused residual/relu/BN outputs
  finalize<<<dim3(131072), 256, 0, stream>>>(x, oH, oW, st, bnh_w, bnh_b,
                                             bnw_w, bnw_b, gamma, out);
}
